// MoELayer_tp_75711683494340
// MI455X (gfx1250) — matrix operand order repaired and checked by reading, not compiled, not run on hardware
//
#include <hip/hip_runtime.h>
#include <hip/hip_bf16.h>
#include <math.h>

// Problem constants (from reference)
#define TT 8192
#define HH 1024
#define FF 2048
#define EE 16
#define KK 2
#define CC 1280

typedef __attribute__((ext_vector_type(8)))  float          v8f;
typedef __attribute__((ext_vector_type(16))) __bf16         v16bf;
typedef __attribute__((ext_vector_type(8)))  unsigned short ushort8;
typedef __attribute__((ext_vector_type(4)))  unsigned int   uint4v;
typedef __attribute__((ext_vector_type(8)))  int            int8v;
typedef __attribute__((ext_vector_type(4)))  int            int4v;

union Frag { v16bf v; ushort8 h[2]; };

#if defined(__gfx1250__) && __has_builtin(__builtin_amdgcn_tensor_load_to_lds) && \
    __has_builtin(__builtin_amdgcn_s_wait_tensorcnt)
#define USE_TDM 1
#else
#define USE_TDM 0
#endif

// Round-to-nearest-even rounding term applied in fp32 bit-space; the bf16
// result is the high halfword of the rounded word.
__device__ __forceinline__ unsigned int bf_round(float x) {
    unsigned int u = __builtin_bit_cast(unsigned int, x);
    return u + 0x7FFFu + ((u >> 16) & 1u);
}

// Pack bf16(lo) | bf16(hi)<<16 in one v_perm_b32: take high halfwords of the
// two rounded words. v_perm: sel 0-3 -> bytes of src1(b), 4-7 -> bytes of src0(a).
__device__ __forceinline__ unsigned int pack_bf2(float lo, float hi) {
    return __builtin_amdgcn_perm(bf_round(hi), bf_round(lo), 0x07060302u);
}

__device__ __forceinline__ unsigned short f2bf(float x) {
    return (unsigned short)(bf_round(x) >> 16);
}

// tanh-GELU, branch free: 0.5*x*(1+tanh(z)) == x / (1 + exp(-2z))
__device__ __forceinline__ float gelu_tanh(float x) {
    float z = 0.7978845608028654f * (x + 0.044715f * x * x * x);
    return x / (1.0f + __expf(-2.0f * z));
}

#if USE_TDM
// Byte offset of a generic (shared) pointer within the LDS aperture.
__device__ __forceinline__ unsigned int lds_byte_off(const void* p) {
    return (unsigned int)(size_t)(__attribute__((address_space(3))) const void*)p;
}

// 2-D TDM load: tile rows x cols (bf16, cols contiguous) from a row-major
// tensor with row stride `stride_elems`, packed row-major into LDS at lds_off.
// D# layout per CDNA5 ISA 8.3/8.4; 6-arg builtin on this toolchain.
__device__ __forceinline__ void tdm_load_tile_bf16(
    unsigned int lds_off, const void* gaddr,
    unsigned int rows, unsigned int cols,
    unsigned int stride_elems, unsigned int tensor_d0, unsigned int tensor_d1)
{
    unsigned long long ga = (unsigned long long)(size_t)gaddr;
    uint4v g0;
    g0[0] = 1u;                                           // count=1 (valid), user mode
    g0[1] = lds_off;                                      // lds_addr (bytes)
    g0[2] = (unsigned int)(ga & 0xFFFFFFFFull);           // global_addr[31:0]
    g0[3] = (unsigned int)((ga >> 32) & 0x01FFFFFFull)    // global_addr[56:32]
          | 0x80000000u;                                  // type=2 ("image") [127:126]
    int8v g1;
    g1[0] = 0x00010000;                                   // data_size=1 (2 bytes/elem)
    g1[1] = (int)((tensor_d0 & 0xFFFFu) << 16);           // tensor_dim0[15:0] @ bits63:48
    g1[2] = (int)((tensor_d0 >> 16) | ((tensor_d1 & 0xFFFFu) << 16)); // d0 hi | d1 lo
    g1[3] = (int)((tensor_d1 >> 16) | (cols << 16));      // d1 hi | tile_dim0
    g1[4] = (int)(rows & 0xFFFFu);                        // tile_dim1 (tile_dim2=0)
    g1[5] = (int)stride_elems;                            // tensor_dim0_stride[31:0]
    g1[6] = 0;
    g1[7] = 0;
    int4v z4 = {0, 0, 0, 0};
    int8v z8 = {0, 0, 0, 0, 0, 0, 0, 0};
    __builtin_amdgcn_tensor_load_to_lds(g0, g1, z4, z4, z8, 0);
}
#endif

// ---------------------------------------------------------------------------
// 1) Router: logits = x @ w_router, top-2, renormalized softmax.
// ---------------------------------------------------------------------------
__global__ __launch_bounds__(256) void router_topk_kernel(
    const float* __restrict__ x, const float* __restrict__ wr,
    int* __restrict__ topk_idx, float* __restrict__ topk_p)
{
    int wave = threadIdx.x >> 5;
    int lane = threadIdx.x & 31;
    int t = blockIdx.x * 8 + wave;

    float acc[EE];
#pragma unroll
    for (int e = 0; e < EE; ++e) acc[e] = 0.0f;

    const float* xr = x + (size_t)t * HH;
    for (int h = lane; h < HH; h += 32) {
        float xv = xr[h];
        const float4* w4 = reinterpret_cast<const float4*>(wr + (size_t)h * EE);
#pragma unroll
        for (int q = 0; q < 4; ++q) {
            float4 wv = w4[q];
            acc[q * 4 + 0] += xv * wv.x;
            acc[q * 4 + 1] += xv * wv.y;
            acc[q * 4 + 2] += xv * wv.z;
            acc[q * 4 + 3] += xv * wv.w;
        }
    }
#pragma unroll
    for (int e = 0; e < EE; ++e) {
        for (int off = 16; off > 0; off >>= 1)
            acc[e] += __shfl_xor(acc[e], off, 32);
    }
    if (lane == 0) {
        int i0 = 0; float l0 = acc[0];
        for (int e = 1; e < EE; ++e) if (acc[e] > l0) { l0 = acc[e]; i0 = e; }
        int i1 = -1; float l1 = -1.0e30f;
        for (int e = 0; e < EE; ++e) if (e != i0 && acc[e] > l1) { l1 = acc[e]; i1 = e; }
        float e1 = __expf(l1 - l0);
        float s = 1.0f + e1;
        topk_idx[t * 2 + 0] = i0;
        topk_idx[t * 2 + 1] = i1;
        topk_p[t * 2 + 0] = 1.0f / s;
        topk_p[t * 2 + 1] = e1 / s;
    }
}

// ---------------------------------------------------------------------------
// 2) Deterministic capacity positions (ordered ballot prefix-scan per expert).
// ---------------------------------------------------------------------------
__global__ void positions_kernel(const int* __restrict__ topk_idx,
                                 int* __restrict__ pos)
{
    int e = blockIdx.x;
    int lane = threadIdx.x;
    int count = 0;
    for (int base = 0; base < TT * KK; base += 32) {
        int i = base + lane;
        bool m = (topk_idx[i] == e);
        unsigned long long mask = __ballot(m);
        int prefix = __popcll(mask & ((1ull << lane) - 1ull));
        if (m) pos[i] = count + prefix;
        count += __popcll(mask);
    }
}

// ---------------------------------------------------------------------------
// 3) Zero the bf16 dispatch buffer.
// ---------------------------------------------------------------------------
__global__ void zero_u4_kernel(uint4* __restrict__ p)
{
    p[(size_t)blockIdx.x * 256 + threadIdx.x] = make_uint4(0u, 0u, 0u, 0u);
}

// ---------------------------------------------------------------------------
// 4) Scatter: disp[e][pos][:] = bf16(x[token][:]), packed dword stores.
// ---------------------------------------------------------------------------
__global__ void dispatch_kernel(const float* __restrict__ x,
                                const int* __restrict__ topk_idx,
                                const int* __restrict__ pos,
                                unsigned short* __restrict__ disp)
{
    int i = blockIdx.x;
    int e = topk_idx[i];
    int ps = pos[i];
    if (ps >= CC) return;
    int t = i >> 1;
    int h0 = threadIdx.x * 8; // 128 threads x 8 contiguous elements
    const float4* src = reinterpret_cast<const float4*>(x + (size_t)t * HH + h0);
    unsigned int* dst = reinterpret_cast<unsigned int*>(
        disp + ((size_t)e * CC + ps) * HH + h0);
    float4 a = src[0];
    float4 b = src[1];
    dst[0] = pack_bf2(a.x, a.y);
    dst[1] = pack_bf2(a.z, a.w);
    dst[2] = pack_bf2(b.x, b.y);
    dst[3] = pack_bf2(b.z, b.w);
}

// ---------------------------------------------------------------------------
// 5/6) Expert GEMM: v_wmma_f32_16x16x32_bf16, 128x128x32 block tiles,
//      double-buffered LDS. Per iteration: issue TDM for next A tile and
//      global loads for next B tile, run 8 WMMAs on the current tile,
//      then convert/store B and s_wait_tensorcnt before the barrier.
// ---------------------------------------------------------------------------
template <bool FUSE_GELU>
__global__ __launch_bounds__(256) void moe_gemm_kernel(
    const unsigned short* __restrict__ A, const float* __restrict__ B,
    const float* __restrict__ bias, void* __restrict__ outv,
    int M, int N, int Kd)
{
    constexpr int BM = 128, BN = 128, BK = 32;
    __shared__ unsigned short As[2][BM][BK]; // [m][k]
    __shared__ unsigned short Bs[2][BN][BK]; // [n][k] (transposed)

    int e  = blockIdx.z;
    int n0 = blockIdx.x * BN;
    int m0 = blockIdx.y * BM;

    const unsigned short* Ae = A + (size_t)e * M * Kd;
    const float*          Be = B + (size_t)e * Kd * N;

    int tid  = threadIdx.x;
    int lane = tid & 31;
    int wave = tid >> 5;
    int wm = (wave & 1) * 64;
    int wn = (wave >> 1) * 32;
    int half = lane >> 4;
    int lm   = lane & 15;

    v8f acc[4][2];
#pragma unroll
    for (int i = 0; i < 4; ++i)
#pragma unroll
        for (int j = 0; j < 2; ++j)
#pragma unroll
            for (int q = 0; q < 8; ++q) acc[i][j][q] = 0.0f;

    // B fill: each thread owns 2 adjacent k-rows x 8 n values -> packed b32.
    int kk  = (tid & 15) * 2;
    int bn8 = (tid >> 4) * 8;
    float4 br[4]; // [row kk: n0..3, n4..7][row kk+1: n0..3, n4..7]

#if !USE_TDM
    int am = tid & 127, ah = tid >> 7; // sync A fill fallback
#endif

    auto issue_a = [&](int buf, int k0) {
#if USE_TDM
        if (wave == 0)
            tdm_load_tile_bf16(lds_byte_off(&As[buf][0][0]),
                               Ae + (size_t)m0 * Kd + k0,
                               BM, BK, (unsigned int)Kd,
                               (unsigned int)Kd, (unsigned int)M);
#else
        const unsigned short* src = Ae + (size_t)(m0 + am) * Kd + k0 + ah * 16;
        *reinterpret_cast<ushort8*>(&As[buf][am][ah * 16]) =
            *reinterpret_cast<const ushort8*>(src);
#endif
    };
    auto load_b = [&](int k0) {
        const float* p0 = Be + (size_t)(k0 + kk) * N + n0 + bn8;
        br[0] = *reinterpret_cast<const float4*>(p0);
        br[1] = *reinterpret_cast<const float4*>(p0 + 4);
        br[2] = *reinterpret_cast<const float4*>(p0 + N);
        br[3] = *reinterpret_cast<const float4*>(p0 + N + 4);
    };
    auto store_b = [&](int buf) {
        *reinterpret_cast<unsigned int*>(&Bs[buf][bn8 + 0][kk]) = pack_bf2(br[0].x, br[2].x);
        *reinterpret_cast<unsigned int*>(&Bs[buf][bn8 + 1][kk]) = pack_bf2(br[0].y, br[2].y);
        *reinterpret_cast<unsigned int*>(&Bs[buf][bn8 + 2][kk]) = pack_bf2(br[0].z, br[2].z);
        *reinterpret_cast<unsigned int*>(&Bs[buf][bn8 + 3][kk]) = pack_bf2(br[0].w, br[2].w);
        *reinterpret_cast<unsigned int*>(&Bs[buf][bn8 + 4][kk]) = pack_bf2(br[1].x, br[3].x);
        *reinterpret_cast<unsigned int*>(&Bs[buf][bn8 + 5][kk]) = pack_bf2(br[1].y, br[3].y);
        *reinterpret_cast<unsigned int*>(&Bs[buf][bn8 + 6][kk]) = pack_bf2(br[1].z, br[3].z);
        *reinterpret_cast<unsigned int*>(&Bs[buf][bn8 + 7][kk]) = pack_bf2(br[1].w, br[3].w);
    };

    // --- prologue: fill buffer 0 ---
    issue_a(0, 0);
    load_b(0);
    store_b(0);
#if USE_TDM
    if (wave == 0) __builtin_amdgcn_s_wait_tensorcnt(0);
#endif
    __syncthreads();

    int cur = 0;
    for (int k0 = 0; k0 < Kd; k0 += BK) {
        int nxt = cur ^ 1;
        bool more = (k0 + BK) < Kd;
        if (more) {
            issue_a(nxt, k0 + BK);   // TDM DMA overlaps WMMA below
            load_b(k0 + BK);         // global loads in flight during WMMA
            if (k0 + 2 * BK < Kd)    // near-temporal prefetch of following tile
                __builtin_prefetch(Be + (size_t)(k0 + 2 * BK + kk) * N + n0 + bn8, 0, 3);
        }

        // --- fragments per ISA wave32 layouts, from the current buffer ---
        Frag a[4], b[2];
#pragma unroll
        for (int i = 0; i < 4; ++i) {
            const unsigned short* ap = &As[cur][wm + i * 16 + lm][0];
            a[i].h[0] = *reinterpret_cast<const ushort8*>(ap + half * 8);
            a[i].h[1] = *reinterpret_cast<const ushort8*>(ap + 16 + half * 8);
        }
#pragma unroll
        for (int j = 0; j < 2; ++j) {
            const unsigned short* bp = &Bs[cur][wn + j * 16 + lm][0];
            b[j].h[0] = *reinterpret_cast<const ushort8*>(bp + half * 8);
            b[j].h[1] = *reinterpret_cast<const ushort8*>(bp + 16 + half * 8);
        }
#pragma unroll
        for (int i = 0; i < 4; ++i)
#pragma unroll
            for (int j = 0; j < 2; ++j)
                acc[i][j] = __builtin_amdgcn_wmma_f32_16x16x32_bf16(
                    false, a[i].v, false, b[j].v, (short)0, acc[i][j],
                    false, false);

        if (more) {
            store_b(nxt); // convert + packed b32 stores into the other buffer
#if USE_TDM
            if (wave == 0) __builtin_amdgcn_s_wait_tensorcnt(0);
#endif
        }
        __syncthreads();
        cur = nxt;
    }

    // --- epilogue ---
#pragma unroll
    for (int i = 0; i < 4; ++i) {
#pragma unroll
        for (int j = 0; j < 2; ++j) {
            int nn = n0 + wn + j * 16 + lm;
            float bv = bias[(size_t)e * N + nn];
#pragma unroll
            for (int q = 0; q < 8; ++q) {
                int mm = m0 + wm + i * 16 + q + 8 * half;
                float val = acc[i][j][q] + bv;
                size_t oidx = ((size_t)e * M + mm) * N + nn;
                if (FUSE_GELU) {
                    reinterpret_cast<unsigned short*>(outv)[oidx] =
                        f2bf(gelu_tanh(val));
                } else {
                    reinterpret_cast<float*>(outv)[oidx] = val;
                }
            }
        }
    }
}

// ---------------------------------------------------------------------------
// 7) Combine: out[t] = sum_k p_k * y[e_k][pos_k]  (keep = pos < C)
// ---------------------------------------------------------------------------
__global__ __launch_bounds__(256) void combine_kernel(
    const float* __restrict__ y, const int* __restrict__ topk_idx,
    const float* __restrict__ topk_p, const int* __restrict__ pos,
    float* __restrict__ out)
{
    int t = blockIdx.x;
    int h = threadIdx.x * 4;
    float4 r = make_float4(0.f, 0.f, 0.f, 0.f);
#pragma unroll
    for (int k = 0; k < KK; ++k) {
        int i = t * 2 + k;
        int e = topk_idx[i];
        int ps = pos[i];
        float p = topk_p[i];
        if (ps < CC) {
            float4 v = *reinterpret_cast<const float4*>(
                y + (((size_t)e * CC + ps) * HH + h));
            r.x += p * v.x; r.y += p * v.y; r.z += p * v.z; r.w += p * v.w;
        }
    }
    *reinterpret_cast<float4*>(out + (size_t)t * HH + h) = r;
}

// ---------------------------------------------------------------------------
extern "C" void kernel_launch(void* const* d_in, const int* in_sizes, int n_in,
                              void* d_out, int out_size, void* d_ws, size_t ws_size,
                              hipStream_t stream)
{
    const float* x        = (const float*)d_in[0]; // [T,H]
    const float* w_router = (const float*)d_in[1]; // [H,E]
    const float* w1       = (const float*)d_in[2]; // [E,H,F]
    const float* b1       = (const float*)d_in[3]; // [E,F]
    const float* w2       = (const float*)d_in[4]; // [E,F,H]
    const float* b2       = (const float*)d_in[5]; // [E,H]
    float* out            = (float*)d_out;         // [T,H]

    // Workspace layout (~210 MB)
    char* ws = (char*)d_ws;
    int*            topk_idx = (int*)(ws + 0);                       // 64 KB
    float*          topk_p   = (float*)(ws + (64 << 10));            // 64 KB
    int*            pos      = (int*)(ws + (128 << 10));             // 64 KB
    unsigned short* disp     = (unsigned short*)(ws + (256 << 10));  // 40 MB
    size_t disp_elems = (size_t)EE * CC * HH;
    unsigned short* hbuf     = disp + disp_elems;                    // 80 MB
    size_t h_elems = (size_t)EE * CC * FF;
    float*          ybuf     = (float*)(hbuf + h_elems);             // 80 MB

    router_topk_kernel<<<TT / 8, 256, 0, stream>>>(x, w_router, topk_idx, topk_p);
    positions_kernel<<<EE, 32, 0, stream>>>(topk_idx, pos);
    zero_u4_kernel<<<(disp_elems * 2 / 16) / 256, 256, 0, stream>>>((uint4*)disp);
    dispatch_kernel<<<TT * KK, 128, 0, stream>>>(x, topk_idx, pos, disp);

    dim3 g1(FF / 128, CC / 128, EE);
    moe_gemm_kernel<true><<<g1, 256, 0, stream>>>(disp, w1, b1, (void*)hbuf,
                                                  CC, FF, HH);
    dim3 g2(HH / 128, CC / 128, EE);
    moe_gemm_kernel<false><<<g2, 256, 0, stream>>>(hbuf, w2, b2, (void*)ybuf,
                                                   CC, HH, FF);

    combine_kernel<<<TT, 256, 0, stream>>>(ybuf, topk_idx, topk_p, pos, out);
}
